// ObjectAttentionBlock_35485019800069
// MI455X (gfx1250) — matrix operand order repaired and checked by reading, not compiled, not run on hardware
//
#include <hip/hip_runtime.h>

// ---------------------------------------------------------------------------
// ObjectAttentionBlock fused for MI455X (gfx1250, wave32, WMMA bf16)
// Each wave owns output-channel strips; one weight (A) fragment is reused by
// 4 pixel tiles.  All 4 B fragments are fetched into registers before the
// 4 WMMAs so LDS latency is batched, not serialized per-WMMA.
// ---------------------------------------------------------------------------
typedef unsigned short u16;
typedef __attribute__((ext_vector_type(16))) __bf16 v16bf;
typedef __attribute__((ext_vector_type(8)))  float  v8f;

static constexpr int Bn  = 8;
static constexpr int Cc  = 512;
static constexpr int KC  = 256;
static constexpr int Mm  = 19;
static constexpr int Mp  = 32;         // padded object count
static constexpr int Pp  = 128 * 128;  // pixels per batch
static constexpr int TP  = 64;         // pixel tile per workgroup (4 x 16)
static constexpr int XT_PITCH = 520;   // halves, 512 + 8 pad
static constexpr int QT_PITCH = 264;   // halves, 256 + 8 pad

// ws layout (in u16 elements)
static constexpr int WS_WP1 = 0;                  // 256*512
static constexpr int WS_WP2 = 131072;             // 256*256
static constexpr int WS_WU  = 196608;             // 512*256
static constexpr int WS_K   = 327680;             // B * 32 * 256   ([b][m][kc])
static constexpr int WS_V   = 393216;             // B * 256 * 32   ([b][kc][m])

__device__ __forceinline__ u16 f2bf(float f) {
    union { float f; unsigned u; } cv; cv.f = f;
    unsigned u = cv.u;
    unsigned r = u + 0x7FFFu + ((u >> 16) & 1u);   // round-to-nearest-even
    return (u16)(r >> 16);
}

union FragAB { v16bf v; uint4 q[2]; };
union PK8    { u16 h[8]; uint4 u4; };

// Multi-tile GEMM: one A fragment (16 x K strip of weights, [M][K] row-major)
// reused against NT B fragments (B stored transposed [N][K] row-major).
// Layouts per CDNA5 ISA:
//   A[lane][j] = A[m=lane&15][16*(j>>3) + 8*(lane>>4) + (j&7)]
//   B[lane][j] = B[k=16*(j>>3) + 8*(lane>>4) + (j&7)][n=lane&15]
// All NT B fragments are loaded into distinct registers before the WMMAs.
template <int K, int NT>
__device__ __forceinline__ void gemm_nt(const u16* __restrict__ A, int apitch,
                                        const u16* __restrict__ Bt, int bpitch,
                                        int mbase, int lane, v8f* acc) {
    const int lm = lane & 15, hi = lane >> 4;
    const u16* ap = A + (size_t)(mbase + lm) * apitch + 8 * hi;
    const u16* bp0 = Bt + (size_t)lm * bpitch + 8 * hi;
#pragma unroll
    for (int k = 0; k < K; k += 32) {
        FragAB a;
        a.q[0] = *(const uint4*)(ap + k);        // j = 0..7
        a.q[1] = *(const uint4*)(ap + k + 16);   // j = 8..15
        FragAB b[NT];
#pragma unroll
        for (int t = 0; t < NT; ++t) {
            const u16* bp = bp0 + (size_t)(t * 16) * bpitch;
            b[t].q[0] = *(const uint4*)(bp + k);     // j = 0..7
            b[t].q[1] = *(const uint4*)(bp + k + 16);
        }
#pragma unroll
        for (int t = 0; t < NT; ++t)
            acc[t] = __builtin_amdgcn_wmma_f32_16x16x32_bf16(
                         false, a.v, false, b[t].v, (short)0, acc[t], false, false);
    }
}

__device__ __forceinline__ void zero4(v8f* acc) {
#pragma unroll
    for (int t = 0; t < 4; ++t)
        acc[t] = (v8f){0.f, 0.f, 0.f, 0.f, 0.f, 0.f, 0.f, 0.f};
}

// ---------------------------------------------------------------------------
// Kernel 1: convert the three big weight matrices to bf16 in ws
// ---------------------------------------------------------------------------
__global__ void wconv_kernel(const float* __restrict__ wp1,
                             const float* __restrict__ wp2,
                             const float* __restrict__ wu,
                             u16* __restrict__ ws) {
    int idx = blockIdx.x * blockDim.x + threadIdx.x;
    if (idx < 131072)       ws[WS_WP1 + idx] = f2bf(wp1[idx]);
    else if (idx < 196608)  ws[idx]          = f2bf(wp2[idx - 131072]);
    else if (idx < 327680)  ws[idx]          = f2bf(wu [idx - 196608]);
}

// ---------------------------------------------------------------------------
// Kernel 2: k / v object projections (tiny: 152 vectors, 2-layer MLP each)
//   kws[b][m][kc]  bf16  (m padded to 32 with zeros)
//   vws[b][kc][m]  bf16  (m padded to 32 with zeros)
// ---------------------------------------------------------------------------
__global__ __launch_bounds__(256)
void kv_kernel(const float* __restrict__ proxy,
               const float* __restrict__ Wo1, const float* __restrict__ Wo2,
               const float* __restrict__ Wd1, const float* __restrict__ Wd2,
               const float* __restrict__ s_o1, const float* __restrict__ b_o1,
               const float* __restrict__ s_o2, const float* __restrict__ b_o2,
               const float* __restrict__ s_d1, const float* __restrict__ b_d1,
               const float* __restrict__ s_d2, const float* __restrict__ b_d2,
               u16* __restrict__ kws, u16* __restrict__ vws) {
    __shared__ float pvec[Cc];
    __shared__ float h1k[KC];
    __shared__ float h1v[KC];
    const int m = blockIdx.x, bb = blockIdx.y, tid = threadIdx.x;
    if (m >= Mm) {                              // zero padding rows/cols
        kws[(size_t)bb * Mp * KC + m * KC + tid] = 0;
        vws[(size_t)bb * KC * Mp + tid * Mp + m] = 0;
        return;
    }
    for (int c = tid; c < Cc; c += 256)
        pvec[c] = proxy[((size_t)bb * Cc + c) * Mm + m];
    __syncthreads();
    {
        float ak = 0.f, av = 0.f;
        for (int c = 0; c < Cc; ++c) {
            float p = pvec[c];
            ak += Wo1[(size_t)tid * Cc + c] * p;
            av += Wd1[(size_t)tid * Cc + c] * p;
        }
        float hk = ak * s_o1[tid] + b_o1[tid];  h1k[tid] = hk > 0.f ? hk : 0.f;
        float hv = av * s_d1[tid] + b_d1[tid];  h1v[tid] = hv > 0.f ? hv : 0.f;
    }
    __syncthreads();
    {
        float ak = 0.f, av = 0.f;
        for (int c = 0; c < KC; ++c) {
            ak += Wo2[(size_t)tid * KC + c] * h1k[c];
            av += Wd2[(size_t)tid * KC + c] * h1v[c];
        }
        float kk = ak * s_o2[tid] + b_o2[tid];  kk = kk > 0.f ? kk : 0.f;
        float vv = av * s_d2[tid] + b_d2[tid];  vv = vv > 0.f ? vv : 0.f;
        kws[(size_t)bb * Mp * KC + m * KC + tid] = f2bf(kk);
        vws[(size_t)bb * KC * Mp + tid * Mp + m] = f2bf(vv);
    }
}

// ---------------------------------------------------------------------------
// Kernel 3: fused pixel path. One workgroup = 64 pixels of one batch.
//   stage1: q1 = relu(s*(Wp1 . x)+b)        [256 x 64]
//   stage2: q2 = relu(s*(Wp2 . q1)+b)       [256 x 64]
//   stage3: sim = (q2^T . k^T) / 16         [64 x 32]
//   stage4: softmax over m (19 valid)
//   stage5: ctx = v . sim^T                 [256 x 64]
//   stage6: out = relu(s*(Wu . ctx)+b)      [512 x 64] -> HBM
// All intermediates stay in LDS (transposed [pixel][channel] bf16, ready as
// WMMA B operands).
// ---------------------------------------------------------------------------
static constexpr int SM_XT   = 0;        // 64*520*2 = 66560 (reused as ctxT)
static constexpr int SM_Q1   = 66560;    // 64*264*2 = 33792
static constexpr int SM_Q2   = 100352;   // 64*264*2 = 33792
static constexpr int SM_SF   = 134144;   // 64*32*4  = 8192
static constexpr int SM_SB   = 142336;   // 64*32*2  = 4096
static constexpr int SM_TOT  = 146432;

__global__ __launch_bounds__(256)
void fused_attn_kernel(const float* __restrict__ x,
                       const u16* __restrict__ wp1, const u16* __restrict__ wp2,
                       const u16* __restrict__ wu,
                       const u16* __restrict__ kws, const u16* __restrict__ vws,
                       const float* __restrict__ s_p1, const float* __restrict__ b_p1,
                       const float* __restrict__ s_p2, const float* __restrict__ b_p2,
                       const float* __restrict__ s_u,  const float* __restrict__ b_u,
                       float* __restrict__ out) {
    __shared__ __align__(16) unsigned char smem[SM_TOT];
    u16*   xT   = (u16*)(smem + SM_XT);    // [64][520] bf16
    u16*   q1T  = (u16*)(smem + SM_Q1);    // [64][264] bf16
    u16*   q2T  = (u16*)(smem + SM_Q2);    // [64][264] bf16
    float* simf = (float*)(smem + SM_SF);  // [64][32]  f32
    u16*   simb = (u16*)(smem + SM_SB);    // [64][32]  bf16
    u16*   ctxT = xT;                      // reuse region: [64][264] bf16

    const int tid  = threadIdx.x;
    const int lane = tid & 31;
    const int wv   = tid >> 5;             // 8 wave32s
    const int lm   = lane & 15, hi = lane >> 4;
    const int bb   = blockIdx.y;
    const int px0  = blockIdx.x * TP;

    // ---- stage 0: load & transpose x tile into LDS (f32 -> bf16) ----------
    {
        const float* xb = x + (size_t)bb * Cc * Pp + px0;
        const int pxl = tid & 63, c0 = tid >> 6;       // 4 channel rows / pass
        for (int c = c0; c < Cc; c += 4)
            xT[pxl * XT_PITCH + c] = f2bf(xb[(size_t)c * Pp + pxl]);
    }
    __syncthreads();

    // ---- stage 1: q1 = relu(s*(Wp1 . x)+b) --------------------------------
    // wave w owns cout tiles {2w, 2w+1}; A fragment reused across 4 px tiles.
    for (int ci = 0; ci < 2; ++ci) {
        const int ct = wv * 2 + ci;
        v8f acc[4]; zero4(acc);
        gemm_nt<Cc, 4>(wp1, Cc, xT, XT_PITCH, ct * 16, lane, acc);
        const int cb = ct * 16 + 8 * hi;
        float sc[8], bi[8];
#pragma unroll
        for (int r = 0; r < 8; ++r) { sc[r] = s_p1[cb + r]; bi[r] = b_p1[cb + r]; }
#pragma unroll
        for (int t = 0; t < 4; ++t) {
            PK8 pk;
#pragma unroll
            for (int r = 0; r < 8; ++r) {
                float v = acc[t][r] * sc[r] + bi[r];
                pk.h[r] = f2bf(v > 0.f ? v : 0.f);
            }
            *(uint4*)(q1T + (t * 16 + lm) * QT_PITCH + cb) = pk.u4;
        }
    }
    __syncthreads();

    // ---- stage 2: q2 = relu(s*(Wp2 . q1)+b) -------------------------------
    for (int ci = 0; ci < 2; ++ci) {
        const int ct = wv * 2 + ci;
        v8f acc[4]; zero4(acc);
        gemm_nt<KC, 4>(wp2, KC, q1T, QT_PITCH, ct * 16, lane, acc);
        const int cb = ct * 16 + 8 * hi;
        float sc[8], bi[8];
#pragma unroll
        for (int r = 0; r < 8; ++r) { sc[r] = s_p2[cb + r]; bi[r] = b_p2[cb + r]; }
#pragma unroll
        for (int t = 0; t < 4; ++t) {
            PK8 pk;
#pragma unroll
            for (int r = 0; r < 8; ++r) {
                float v = acc[t][r] * sc[r] + bi[r];
                pk.h[r] = f2bf(v > 0.f ? v : 0.f);
            }
            *(uint4*)(q2T + (t * 16 + lm) * QT_PITCH + cb) = pk.u4;
        }
    }
    __syncthreads();

    // ---- stage 3: sim[px][m] = (q2^T . k^T)[px][m] * kc^-0.5 --------------
    // 4 px tiles x 2 m tiles = 8 output tiles, one per wave.
    {
        const int pxt = (wv & 3) * 16;
        const int mt  = (wv >> 2) * 16;
        const u16* kb = kws + (size_t)bb * Mp * KC;      // [m][kc]
        const u16* ap = q2T + (size_t)(pxt + lm) * QT_PITCH + 8 * hi;  // A = q2^T
        const u16* bp = kb + (size_t)(mt + lm) * KC + 8 * hi;
        v8f acc = {0.f, 0.f, 0.f, 0.f, 0.f, 0.f, 0.f, 0.f};
#pragma unroll
        for (int k = 0; k < KC; k += 32) {
            FragAB a, b;
            a.q[0] = *(const uint4*)(ap + k);
            a.q[1] = *(const uint4*)(ap + k + 16);
            b.q[0] = *(const uint4*)(bp + k);
            b.q[1] = *(const uint4*)(bp + k + 16);
            acc = __builtin_amdgcn_wmma_f32_16x16x32_bf16(
                      false, a.v, false, b.v, (short)0, acc, false, false);
        }
#pragma unroll
        for (int r = 0; r < 8; ++r)
            simf[(pxt + r + 8 * hi) * Mp + mt + lm] = acc[r] * 0.0625f;
    }
    __syncthreads();

    // ---- stage 4: softmax over 19 valid objects ---------------------------
    if (tid < TP) {
        float* sp = simf + tid * Mp;
        float mx = sp[0];
        for (int m = 1; m < Mm; ++m) mx = sp[m] > mx ? sp[m] : mx;
        float sum = 0.f;
        for (int m = 0; m < Mm; ++m) { float e = __expf(sp[m] - mx); sp[m] = e; sum += e; }
        const float inv = 1.f / sum;
        u16* sb = simb + tid * Mp;
        for (int m = 0; m < Mm; ++m) sb[m] = f2bf(sp[m] * inv);
        for (int m = Mm; m < Mp; ++m) sb[m] = 0;
    }
    __syncthreads();

    // ---- stage 5: ctx = v . sim^T  (K = 32, one WMMA per tile) ------------
    for (int ci = 0; ci < 2; ++ci) {
        const int ct = wv * 2 + ci;
        const u16* vb = vws + (size_t)bb * KC * Mp;      // [kc][32]
        v8f acc[4]; zero4(acc);
        gemm_nt<Mp, 4>(vb, Mp, simb, Mp, ct * 16, lane, acc);
        const int cb = ct * 16 + 8 * hi;
#pragma unroll
        for (int t = 0; t < 4; ++t) {
            PK8 pk;
#pragma unroll
            for (int r = 0; r < 8; ++r) pk.h[r] = f2bf(acc[t][r]);
            *(uint4*)(ctxT + (t * 16 + lm) * QT_PITCH + cb) = pk.u4;
        }
    }
    __syncthreads();

    // ---- stage 6: out = relu(s*(Wu . ctx)+b) -> global --------------------
    // wave w owns cout tiles {4w .. 4w+3} over C=512.
    for (int ci = 0; ci < 4; ++ci) {
        const int ct = wv * 4 + ci;
        v8f acc[4]; zero4(acc);
        gemm_nt<KC, 4>(wu, KC, ctxT, QT_PITCH, ct * 16, lane, acc);
        const int cb = ct * 16 + 8 * hi;
        float sc[8], bi[8];
#pragma unroll
        for (int r = 0; r < 8; ++r) { sc[r] = s_u[cb + r]; bi[r] = b_u[cb + r]; }
#pragma unroll
        for (int t = 0; t < 4; ++t) {
            const int px = px0 + t * 16 + lm;
#pragma unroll
            for (int r = 0; r < 8; ++r) {
                float v = acc[t][r] * sc[r] + bi[r];
                out[((size_t)bb * Cc + cb + r) * Pp + px] = v > 0.f ? v : 0.f;
            }
        }
    }
}

// ---------------------------------------------------------------------------
extern "C" void kernel_launch(void* const* d_in, const int* in_sizes, int n_in,
                              void* d_out, int out_size, void* d_ws, size_t ws_size,
                              hipStream_t stream) {
    (void)in_sizes; (void)n_in; (void)out_size; (void)ws_size;
    const float* x     = (const float*)d_in[0];
    const float* proxy = (const float*)d_in[1];
    const float* W_p1  = (const float*)d_in[2];
    const float* W_p2  = (const float*)d_in[3];
    const float* W_o1  = (const float*)d_in[4];
    const float* W_o2  = (const float*)d_in[5];
    const float* W_d1  = (const float*)d_in[6];
    const float* W_d2  = (const float*)d_in[7];
    const float* W_u   = (const float*)d_in[8];
    const float* s_p1  = (const float*)d_in[9];
    const float* b_p1  = (const float*)d_in[10];
    const float* s_p2  = (const float*)d_in[11];
    const float* b_p2  = (const float*)d_in[12];
    const float* s_o1  = (const float*)d_in[13];
    const float* b_o1  = (const float*)d_in[14];
    const float* s_o2  = (const float*)d_in[15];
    const float* b_o2  = (const float*)d_in[16];
    const float* s_d1  = (const float*)d_in[17];
    const float* b_d1  = (const float*)d_in[18];
    const float* s_d2  = (const float*)d_in[19];
    const float* b_d2  = (const float*)d_in[20];
    const float* s_u   = (const float*)d_in[21];
    const float* b_u   = (const float*)d_in[22];
    float* out = (float*)d_out;
    u16*   ws  = (u16*)d_ws;

    wconv_kernel<<<(327680 + 255) / 256, 256, 0, stream>>>(W_p1, W_p2, W_u, ws);

    kv_kernel<<<dim3(Mp, Bn), 256, 0, stream>>>(
        proxy, W_o1, W_o2, W_d1, W_d2,
        s_o1, b_o1, s_o2, b_o2, s_d1, b_d1, s_d2, b_d2,
        ws + WS_K, ws + WS_V);

    fused_attn_kernel<<<dim3(Pp / TP, Bn), 256, 0, stream>>>(
        x, ws + WS_WP1, ws + WS_WP2, ws + WS_WU, ws + WS_K, ws + WS_V,
        s_p1, b_p1, s_p2, b_p2, s_u, b_u, out);
}
